// Decoder_88055419502973
// MI455X (gfx1250) — matrix operand order repaired and checked by reading, not compiled, not run on hardware
//
#include <hip/hip_runtime.h>
#include <hip/hip_bf16.h>
#include <math.h>

typedef __attribute__((ext_vector_type(16))) __bf16    v16bf;
typedef __attribute__((ext_vector_type(8)))  __bf16    v8bf;
typedef __attribute__((ext_vector_type(8)))  _Float16  v8h;
typedef __attribute__((ext_vector_type(8)))  float     v8f;

#define BB   512
#define NN   500
#define NP   512   // N padded to 32 tiles of 16
#define DD   128
#define HH   8
#define DHH  16
#define TT   50
#define NEGV (-1e9f)

static __device__ __forceinline__ v8f wmma_bf16(v16bf a, v16bf b, v8f c) {
  return __builtin_amdgcn_wmma_f32_16x16x32_bf16(false, a, false, b, (short)0, c, false, false);
}
static __device__ __forceinline__ v16bf cat8(v8bf lo, v8bf hi) {
  return __builtin_shufflevector(lo, hi, 0, 1, 2, 3, 4, 5, 6, 7, 8, 9, 10, 11, 12, 13, 14, 15);
}

// ---------------------------------------------------------------------------
// graph_ctx[b,d] = (mean_n emb[b,n,:]) @ W_fixed
// ---------------------------------------------------------------------------
__global__ __launch_bounds__(128) void ctx_kernel(const float* __restrict__ emb,
                                                  const float* __restrict__ Wf,
                                                  float* __restrict__ gctx) {
  const int b = blockIdx.x;
  const int d = threadIdx.x;
  __shared__ float mlds[DD];
  float s = 0.f;
  const float* eb = emb + (size_t)b * NN * DD + d;
  for (int n = 0; n < NN; ++n) s += eb[(size_t)n * DD];
  mlds[d] = s * (1.0f / (float)NN);
  __syncthreads();
  float acc = 0.f;
  for (int k = 0; k < DD; ++k) acc += mlds[k] * Wf[(size_t)k * DD + d];
  gctx[(size_t)b * DD + d] = acc;
}

// Wt[d][k] = bf16(W_out[k][d])   (transposed for phase-3 B fragments)
__global__ __launch_bounds__(128) void wout_kernel(const float* __restrict__ Wo,
                                                   __bf16* __restrict__ Wt) {
  const int d = blockIdx.x, k = threadIdx.x;
  Wt[(size_t)d * DD + k] = (__bf16)Wo[(size_t)k * DD + d];
}

// ---------------------------------------------------------------------------
// proj = emb @ W_node (bf16 WMMA, f32 accum). A rows are single-use ->
// direct global->fragment loads (no LDS). Only W tile staged in LDS (full K).
//   gk[b][h][n][dh], gv[b][h][dh][n] (transposed), lk[b][n][d]; pads zero.
// ---------------------------------------------------------------------------
__global__ __launch_bounds__(256) void proj_kernel(const float* __restrict__ emb,
                                                   const float* __restrict__ Wn,
                                                   __bf16* __restrict__ gk,
                                                   __bf16* __restrict__ gv,
                                                   __bf16* __restrict__ lko) {
  const int b    = blockIdx.x >> 2;
  const int nblk = (blockIdx.x & 3) << 7;
  const int c0   = blockIdx.y * 64;

  __shared__ alignas(16) __bf16 Bs[64][136];      // [col][k], padded rows
  __shared__ alignas(16) __bf16 tbuf[8][16][24];  // per-wave transpose staging

  const int tid  = threadIdx.x;
  const int wave = tid >> 5;
  const int lane = tid & 31;
  const int g    = lane >> 4;
  const int m    = lane & 15;

  for (int i = tid; i < 64 * 128; i += 256) {
    const int c = i & 63, k = i >> 6;
    Bs[c][k] = (__bf16)Wn[(size_t)k * 384 + c0 + c];
  }
  __syncthreads();

  const int nrow = nblk + wave * 16 + m;
  const bool valid = nrow < NN;
  const float* arow = emb + ((size_t)b * NN + (valid ? nrow : 0)) * DD;

  v8f cacc[4];
  cacc[0] = {}; cacc[1] = {}; cacc[2] = {}; cacc[3] = {};

#pragma unroll
  for (int kc = 0; kc < 4; ++kc) {
    v16bf a;
    if (valid) {
      const float4 f0 = *(const float4*)(arow + kc * 32 + 8 * g);
      const float4 f1 = *(const float4*)(arow + kc * 32 + 8 * g + 4);
      const float4 f2 = *(const float4*)(arow + kc * 32 + 16 + 8 * g);
      const float4 f3 = *(const float4*)(arow + kc * 32 + 16 + 8 * g + 4);
      a[0] = (__bf16)f0.x; a[1] = (__bf16)f0.y; a[2] = (__bf16)f0.z; a[3] = (__bf16)f0.w;
      a[4] = (__bf16)f1.x; a[5] = (__bf16)f1.y; a[6] = (__bf16)f1.z; a[7] = (__bf16)f1.w;
      a[8] = (__bf16)f2.x; a[9] = (__bf16)f2.y; a[10] = (__bf16)f2.z; a[11] = (__bf16)f2.w;
      a[12] = (__bf16)f3.x; a[13] = (__bf16)f3.y; a[14] = (__bf16)f3.z; a[15] = (__bf16)f3.w;
    } else {
      const v8bf z = {};
      a = cat8(z, z);
    }
#pragma unroll
    for (int ct = 0; ct < 4; ++ct) {
      const v16bf bb = cat8(*(const v8bf*)&Bs[ct * 16 + m][kc * 32 + 8 * g],
                            *(const v8bf*)&Bs[ct * 16 + m][kc * 32 + 16 + 8 * g]);
      cacc[ct] = wmma_bf16(a, bb, cacc[ct]);
    }
  }

  if (c0 < 128) {            // gk[b][h][n][dh]
#pragma unroll
    for (int ct = 0; ct < 4; ++ct) {
      const int h = (c0 + ct * 16) >> 4;
#pragma unroll
      for (int j = 0; j < 8; ++j) tbuf[wave][8 * g + j][m] = (__bf16)cacc[ct][j];
      const v8bf out = *(const v8bf*)&tbuf[wave][m][8 * g];
      *(v8bf*)&gk[(((size_t)b * HH + h) * NP + (nblk + wave * 16 + m)) * DHH + 8 * g] = out;
    }
  } else if (c0 < 256) {     // gv[b][h][dh][n]
#pragma unroll
    for (int ct = 0; ct < 4; ++ct) {
      const int h = (c0 - 128 + ct * 16) >> 4;
      v8bf out;
#pragma unroll
      for (int j = 0; j < 8; ++j) out[j] = (__bf16)cacc[ct][j];
      *(v8bf*)&gv[(((size_t)b * HH + h) * DHH + m) * NP + (nblk + wave * 16 + 8 * g)] = out;
    }
  } else {                   // lk[b][n][128]
#pragma unroll
    for (int ct = 0; ct < 4; ++ct) {
      const int cb = c0 - 256 + ct * 16;
#pragma unroll
      for (int j = 0; j < 8; ++j) tbuf[wave][8 * g + j][m] = (__bf16)cacc[ct][j];
      const v8bf out = *(const v8bf*)&tbuf[wave][m][8 * g];
      *(v8bf*)&lko[((size_t)b * NP + (nblk + wave * 16 + m)) * DD + cb + 8 * g] = out;
    }
  }
}

// ---------------------------------------------------------------------------
// Step kernel: two blocks per b (alternate t-chunks); 16 timesteps batched into WMMA columns.
// ---------------------------------------------------------------------------
__global__ __launch_bounds__(256) void step_kernel(const float* __restrict__ gctx,
                                                   const float* __restrict__ sctx,
                                                   const unsigned char* __restrict__ amask,
                                                   const __bf16* __restrict__ Wt,
                                                   const __bf16* __restrict__ gk,
                                                   const __bf16* __restrict__ gv,
                                                   const __bf16* __restrict__ lk,
                                                   float* __restrict__ logp_out,
                                                   float* __restrict__ act_out) {
  const int b    = blockIdx.x;
  const int tid  = threadIdx.x;
  const int wave = tid >> 5;
  const int lane = tid & 31;
  const int g    = lane >> 4;
  const int m    = lane & 15;
  const int h    = wave;

  __shared__ alignas(16) unsigned short scbuf[HH][16][520];   // f16 scores -> bf16 attn
  __shared__ alignas(16) __bf16        qb[16][136];
  __shared__ alignas(16) __bf16        glimb[16][136];
  __shared__ alignas(16) __bf16        gpb[16][136];
  __shared__ alignas(16) unsigned char mask8[16][512];
  float (*lgA)[516] = (float (*)[516])&scbuf[0][0][0];        // [16][516] overlay

  const __bf16* Kh = gk + ((size_t)b * HH + h) * NP * DHH;
  const __bf16* Vt = gv + (((size_t)b * HH + h) * DHH + m) * NP;
  const __bf16* Lb = lk + (size_t)b * NP * DD;
  const v8bf zero8 = {};

  for (int tc = blockIdx.y; tc < 4; tc += 2) {
    // ---- stage q (scaled) and mask chunk (shift-only indexing) ----
    for (int i = tid; i < 16 * DD; i += 256) {
      const int tl = i >> 7, d = i & 127;
      const int tg = min(tc * 16 + tl, TT - 1);
      const float v = gctx[(size_t)b * DD + d] + sctx[((size_t)tg * BB + b) * DD + d];
      qb[tl][d] = (__bf16)(v * 0.25f);   // fold 1/sqrt(16)
    }
    for (int i = tid; i < 16 * 128; i += 256) {
      const int tl = i >> 7, w = i & 127;
      if (w < 125) {
        const int tg = min(tc * 16 + tl, TT - 1);
        ((unsigned int*)&mask8[tl][0])[w] =
            ((const unsigned int*)(amask + ((size_t)tg * BB + b) * NN))[w];
      }
    }
    __syncthreads();

    // ---- phase 1: scores[n][t16] per head ----
    const v16bf bq = cat8(*(const v8bf*)&qb[m][h * DHH + 8 * g], zero8);
    for (int nt = 0; nt < 32; ++nt) {
      const __bf16* krow = Kh + ((size_t)(nt * 16 + m)) * DHH + 8 * g;
      __builtin_prefetch(Kh + ((size_t)((nt + 1) * 16 + m)) * DHH, 0, 1);
      const v16bf a = cat8(*(const v8bf*)krow, zero8);
      v8f cs = {};
      cs = wmma_bf16(a, bq, cs);
      v8h hh;
#pragma unroll
      for (int j = 0; j < 8; ++j) hh[j] = (_Float16)cs[j];
      *(v8h*)&scbuf[h][m][nt * 16 + 8 * g] = hh;
    }

    // ---- per-wave masked softmax (vectorized mask read) ----
    for (int tt = 0; tt < 16; ++tt) {
      const int nb = lane * 16;
      const v8h s0 = *(const v8h*)&scbuf[h][tt][nb];
      const v8h s1 = *(const v8h*)&scbuf[h][tt][nb + 8];
      const uint4 mw = *(const uint4*)&mask8[tt][nb];
      const unsigned wds[4] = {mw.x, mw.y, mw.z, mw.w};
      float v[16];
#pragma unroll
      for (int i = 0; i < 8; ++i) { v[i] = (float)s0[i]; v[8 + i] = (float)s1[i]; }
      float mx = -INFINITY;
#pragma unroll
      for (int i = 0; i < 16; ++i) {
        const bool ok = ((nb + i) < NN) && (((wds[i >> 2] >> (8 * (i & 3))) & 0xffu) != 0u);
        v[i] = ok ? v[i] : NEGV;
        mx = fmaxf(mx, v[i]);
      }
#pragma unroll
      for (int off = 16; off >= 1; off >>= 1) mx = fmaxf(mx, __shfl_xor(mx, off, 32));
      float sum = 0.f;
#pragma unroll
      for (int i = 0; i < 16; ++i) { v[i] = __expf(v[i] - mx); sum += v[i]; }
#pragma unroll
      for (int off = 16; off >= 1; off >>= 1) sum += __shfl_xor(sum, off, 32);
      const float inv = 1.0f / sum;
      v8bf a0, a1;
#pragma unroll
      for (int i = 0; i < 8; ++i) { a0[i] = (__bf16)(v[i] * inv); a1[i] = (__bf16)(v[8 + i] * inv); }
      *(v8bf*)&scbuf[h][tt][nb] = a0;
      *(v8bf*)&scbuf[h][tt][nb + 8] = a1;
    }

    // ---- phase 2: glimpse D[t][d] per head ----
    v8f ga = {};
    for (int ch = 0; ch < 16; ++ch) {
      const int n0 = ch * 32;
      const __bf16* arow = (const __bf16*)&scbuf[h][m][0];
      const v16bf a  = cat8(*(const v8bf*)&arow[n0 + 8 * g],
                            *(const v8bf*)&arow[n0 + 16 + 8 * g]);
      const v16bf bv = cat8(*(const v8bf*)&Vt[n0 + 8 * g],
                            *(const v8bf*)&Vt[n0 + 16 + 8 * g]);
      ga = wmma_bf16(a, bv, ga);
    }
#pragma unroll
    for (int j = 0; j < 8; ++j) glimb[8 * g + j][h * DHH + m] = (__bf16)ga[j];
    __syncthreads();

    // ---- phase 3: gp = glimpse @ W_out (wave = 16-wide d-tile) ----
    v8f gp = {};
#pragma unroll
    for (int kc = 0; kc < 4; ++kc) {
      const v16bf a  = cat8(*(const v8bf*)&glimb[m][kc * 32 + 8 * g],
                            *(const v8bf*)&glimb[m][kc * 32 + 16 + 8 * g]);
      const __bf16* wrow = Wt + (size_t)(wave * 16 + m) * DD + kc * 32;
      const v16bf bw = cat8(*(const v8bf*)&wrow[8 * g],
                            *(const v8bf*)&wrow[16 + 8 * g]);
      gp = wmma_bf16(a, bw, gp);
    }
#pragma unroll
    for (int j = 0; j < 8; ++j) gpb[8 * g + j][wave * 16 + m] = (__bf16)gp[j];
    __syncthreads();

    // ---- phase 4: logits D[n][t16]; wave handles 4 n-tiles ----
    for (int s4 = 0; s4 < 4; ++s4) {
      const int n0 = (h + 8 * s4) * 16;
      v8f la = {};
#pragma unroll
      for (int kc = 0; kc < 4; ++kc) {
        const __bf16* lrow = Lb + (size_t)(n0 + m) * DD + kc * 32;
        const v16bf a  = cat8(*(const v8bf*)(lrow + 8 * g),
                              *(const v8bf*)(lrow + 16 + 8 * g));
        const v16bf bb = cat8(*(const v8bf*)&gpb[m][kc * 32 + 8 * g],
                              *(const v8bf*)&gpb[m][kc * 32 + 16 + 8 * g]);
        la = wmma_bf16(a, bb, la);
      }
      float4 o0, o1;
      o0.x = la[0]; o0.y = la[1]; o0.z = la[2]; o0.w = la[3];
      o1.x = la[4]; o1.y = la[5]; o1.z = la[6]; o1.w = la[7];
      *(float4*)&lgA[m][n0 + 8 * g] = o0;
      *(float4*)&lgA[m][n0 + 8 * g + 4] = o1;
    }
    __syncthreads();

    // ---- phase 5: tanh clip, mask, log_softmax, argmax; wave -> 2 t's ----
#pragma unroll
    for (int half = 0; half < 2; ++half) {
      const int tl = h + 8 * half;
      const int tg = tc * 16 + tl;
      const int nb = lane * 16;
      float lg[16];
#pragma unroll
      for (int q4 = 0; q4 < 4; ++q4) {
        const float4 f = *(const float4*)&lgA[tl][nb + 4 * q4];
        lg[4 * q4 + 0] = f.x; lg[4 * q4 + 1] = f.y; lg[4 * q4 + 2] = f.z; lg[4 * q4 + 3] = f.w;
      }
      const uint4 mw = *(const uint4*)&mask8[tl][nb];
      const unsigned wds[4] = {mw.x, mw.y, mw.z, mw.w};
      float mx = -INFINITY;
#pragma unroll
      for (int i = 0; i < 16; ++i) {
        float t = tanhf(lg[i] * 0.0883883476f) * 10.0f;
        const bool ok = ((nb + i) < NN) && (((wds[i >> 2] >> (8 * (i & 3))) & 0xffu) != 0u);
        lg[i] = ok ? t : NEGV;
        mx = fmaxf(mx, lg[i]);
      }
#pragma unroll
      for (int off = 16; off >= 1; off >>= 1) mx = fmaxf(mx, __shfl_xor(mx, off, 32));
      float sum = 0.f;
      float bestv = -INFINITY; int besti = nb;
#pragma unroll
      for (int i = 0; i < 16; ++i) {
        sum += __expf(lg[i] - mx);
        if (lg[i] > bestv) { bestv = lg[i]; besti = nb + i; }
      }
#pragma unroll
      for (int off = 16; off >= 1; off >>= 1) sum += __shfl_xor(sum, off, 32);
#pragma unroll
      for (int off = 16; off >= 1; off >>= 1) {
        const float ov = __shfl_xor(bestv, off, 32);
        const int   oi = __shfl_xor(besti, off, 32);
        if (ov > bestv || (ov == bestv && oi < besti)) { bestv = ov; besti = oi; }
      }
      if (tg < TT) {
        const float lse = mx + logf(sum);
        float* orow = logp_out + ((size_t)b * TT + tg) * NN + nb;
        const int nfull = (lane < 31) ? 4 : 1;   // lane31: n=496..499 only
#pragma unroll
        for (int q4 = 0; q4 < 4; ++q4) {
          if (q4 < nfull) {
            float4 o;
            o.x = lg[4 * q4 + 0] - lse; o.y = lg[4 * q4 + 1] - lse;
            o.z = lg[4 * q4 + 2] - lse; o.w = lg[4 * q4 + 3] - lse;
            *(float4*)(orow + 4 * q4) = o;
          }
        }
        if (lane == 0) act_out[(size_t)b * TT + tg] = (float)besti;
      }
    }
    __syncthreads();
  }
}

// ---------------------------------------------------------------------------
extern "C" void kernel_launch(void* const* d_in, const int* in_sizes, int n_in,
                              void* d_out, int out_size, void* d_ws, size_t ws_size,
                              hipStream_t stream) {
  (void)in_sizes; (void)n_in; (void)out_size; (void)ws_size;
  const float* emb  = (const float*)d_in[0];
  const float* Wn   = (const float*)d_in[1];
  const float* Wf   = (const float*)d_in[2];
  const float* Wo   = (const float*)d_in[3];
  const float* sctx = (const float*)d_in[4];
  const unsigned char* amask = (const unsigned char*)d_in[5];

  char* ws = (char*)d_ws;
  const size_t SZ_GK = (size_t)BB * HH * NP * DHH * sizeof(__bf16); // 64MB
  const size_t SZ_LK = (size_t)BB * NP * DD * sizeof(__bf16);       // 64MB
  __bf16* gk   = (__bf16*)(ws);
  __bf16* gv   = (__bf16*)(ws + SZ_GK);
  __bf16* lkw  = (__bf16*)(ws + 2 * SZ_GK);
  float*  gctx = (float*)(ws + 2 * SZ_GK + SZ_LK);
  __bf16* wt   = (__bf16*)(ws + 2 * SZ_GK + SZ_LK + (size_t)BB * DD * sizeof(float));

  float* outp = (float*)d_out;
  float* acts = outp + (size_t)BB * TT * NN;

  ctx_kernel<<<dim3(BB), dim3(128), 0, stream>>>(emb, Wf, gctx);
  wout_kernel<<<dim3(DD), dim3(DD), 0, stream>>>(Wo, wt);
  proj_kernel<<<dim3(BB * 4, 6), dim3(256), 0, stream>>>(emb, Wn, gk, gv, lkw);
  step_kernel<<<dim3(BB, 2), dim3(256), 0, stream>>>(gctx, sctx, amask, wt,
                                                     gk, gv, lkw, outp, acts);
}
